// GNNModel_12730283066002
// MI455X (gfx1250) — compile-verified, matrix operand order checked by reading, not run on hardware
//
#include <hip/hip_runtime.h>
#include <hip/hip_bf16.h>

// ---------------- model constants ----------------
#define D_MODEL   64
#define N_HEADS   4
#define HEAD_DIM  16
#define N_CENT    32
#define ALPHA_F   0.05f
#define LEAKY_F   0.2f
#define RBF_G     10.0f
#define WPAD      72        // bf16 LDS row stride (bank-friendly, 16B aligned chunks)

typedef __attribute__((ext_vector_type(16))) __bf16 v16bf;
typedef __attribute__((ext_vector_type(8)))  __bf16 v8bf;
typedef __attribute__((ext_vector_type(8)))  float  v8f;

// ---- order-preserving float->uint key for atomic max over signed floats ----
__device__ __forceinline__ unsigned f2key(float f) {
    unsigned b = __float_as_uint(f);
    return (b & 0x80000000u) ? ~b : (b | 0x80000000u);
}
__device__ __forceinline__ float key2f(unsigned k) {
    unsigned b = (k & 0x80000000u) ? (k & 0x7FFFFFFFu) : ~k;
    return __uint_as_float(b);
}

// ================= WMMA node GEMM helpers (16x16x32 bf16, f32 acc) =========
// A fragment: 16 rows x 32 K, ISA layout: lanes 0-15 -> row=lane, K 0..7 /16..23
//             lanes 16-31 -> row=lane-16, K 8..15 / 24..31
__device__ __forceinline__ v16bf load_a_frag(const float* __restrict__ X,
                                             int row0, int kb, int lane) {
    const int r    = lane & 15;
    const int koff = kb * 32 + ((lane < 16) ? 0 : 8);
    const float* p = X + (size_t)(row0 + r) * D_MODEL + koff;
    v16bf a;
#pragma unroll
    for (int j = 0; j < 8; ++j) a[j]     = (__bf16)p[j];
#pragma unroll
    for (int j = 0; j < 8; ++j) a[8 + j] = (__bf16)p[16 + j];
    return a;
}

// B fragment from LDS-resident bf16 W^T (padded stride WPAD): lw[col*WPAD + k].
// Two aligned 16B chunks -> 2x ds_load_b128, no conversions in the hot loop.
__device__ __forceinline__ v16bf load_b_frag(const __bf16* lw, int nb, int kb, int lane) {
    const int n    = lane & 15;
    const int col  = nb * 16 + n;
    const int koff = kb * 32 + ((lane < 16) ? 0 : 8);
    const __bf16* p = lw + col * WPAD + koff;
    union { v16bf v; v8bf h[2]; } u;
    u.h[0] = *(const v8bf*)(p);
    u.h[1] = *(const v8bf*)(p + 16);
    return u.v;
}

// D 16x16 f32: VGPR i holds M=i (lanes 0-15, N=lane) / M=8+i (lanes 16-31).
// Single base pointer; row stride folds into the 24-bit store IOFFSET.
__device__ __forceinline__ void store_d_frag(float* __restrict__ Y,
                                             int row0, int nb, int lane, v8f d) {
    const int n     = lane & 15;
    const int mbase = (lane < 16) ? 0 : 8;
    float* p = Y + (size_t)(row0 + mbase) * D_MODEL + (nb * 16 + n);
#pragma unroll
    for (int i = 0; i < 8; ++i) p[i * D_MODEL] = d[i];
}

// ---- fused q,k,v = h @ {Wq,Wk,Wv}; one 16-row tile per wave, grid-stride ----
__global__ __launch_bounds__(256) void gemm3_kernel(
    const float* __restrict__ X, int ntiles,
    const float* __restrict__ W0, const float* __restrict__ W1,
    const float* __restrict__ W2,
    float* __restrict__ Y0, float* __restrict__ Y1, float* __restrict__ Y2) {
    __shared__ __bf16 lw[3][64 * WPAD];
#pragma unroll
    for (int w = 0; w < 3; ++w) {
        const float* Wsrc = (w == 0) ? W0 : (w == 1) ? W1 : W2;
        for (int idx = threadIdx.x; idx < 64 * 64; idx += 256) {
            int k2 = idx >> 6, col = idx & 63;
            lw[w][col * WPAD + k2] = (__bf16)Wsrc[k2 * 64 + col]; // transpose+cvt once
        }
    }
    __syncthreads();

    const int lane = threadIdx.x & 31;
    const int wave = threadIdx.x >> 5;
    for (int tile = blockIdx.x * 8 + wave; tile < ntiles; tile += gridDim.x * 8) {
        const int row0 = tile * 16;
        v16bf a0 = load_a_frag(X, row0, 0, lane);
        v16bf a1 = load_a_frag(X, row0, 1, lane);
#pragma unroll
        for (int w = 0; w < 3; ++w) {
            float* Yd = (w == 0) ? Y0 : (w == 1) ? Y1 : Y2;
#pragma unroll
            for (int nb = 0; nb < 4; ++nb) {
                v16bf b0 = load_b_frag(&lw[w][0], nb, 0, lane);
                v16bf b1 = load_b_frag(&lw[w][0], nb, 1, lane);
                v8f c = {};
                c = __builtin_amdgcn_wmma_f32_16x16x32_bf16(false, a0, false, b0,
                                                            (short)0, c, false, false);
                c = __builtin_amdgcn_wmma_f32_16x16x32_bf16(false, a1, false, b1,
                                                            (short)0, c, false, false);
                store_d_frag(Yd, row0, nb, lane, c);
            }
        }
    }
}

// ---- single-weight GEMM: Y = X @ W (output projection) ----
__global__ __launch_bounds__(256) void gemm1_kernel(
    const float* __restrict__ X, int ntiles,
    const float* __restrict__ W, float* __restrict__ Y) {
    __shared__ __bf16 lw[64 * WPAD];
    for (int idx = threadIdx.x; idx < 64 * 64; idx += 256) {
        int k2 = idx >> 6, col = idx & 63;
        lw[col * WPAD + k2] = (__bf16)W[k2 * 64 + col];
    }
    __syncthreads();

    const int lane = threadIdx.x & 31;
    const int wave = threadIdx.x >> 5;
    for (int tile = blockIdx.x * 8 + wave; tile < ntiles; tile += gridDim.x * 8) {
        const int row0 = tile * 16;
        v16bf a0 = load_a_frag(X, row0, 0, lane);
        v16bf a1 = load_a_frag(X, row0, 1, lane);
#pragma unroll
        for (int nb = 0; nb < 4; ++nb) {
            v16bf b0 = load_b_frag(lw, nb, 0, lane);
            v16bf b1 = load_b_frag(lw, nb, 1, lane);
            v8f c = {};
            c = __builtin_amdgcn_wmma_f32_16x16x32_bf16(false, a0, false, b0,
                                                        (short)0, c, false, false);
            c = __builtin_amdgcn_wmma_f32_16x16x32_bf16(false, a1, false, b1,
                                                        (short)0, c, false, false);
            store_d_frag(Y, row0, nb, lane, c);
        }
    }
}

// ---- WrP = rbf_W @ Wr (32x64), brP = rbf_b @ Wr (64); one block ----
__global__ void wrp_kernel(const float* __restrict__ rbf_W,
                           const float* __restrict__ rbf_b,
                           const float* __restrict__ Wr,
                           float* __restrict__ WrP, float* __restrict__ brP) {
    for (int idx = threadIdx.x; idx < N_CENT * D_MODEL; idx += 256) {
        int c = idx >> 6, j = idx & 63;
        float s = 0.f;
        for (int k2 = 0; k2 < 64; ++k2) s += rbf_W[c * 64 + k2] * Wr[k2 * 64 + j];
        WrP[idx] = s;
    }
    for (int j = threadIdx.x; j < D_MODEL; j += 256) {
        float s = 0.f;
        for (int k2 = 0; k2 < 64; ++k2) s += rbf_b[k2] * Wr[k2 * 64 + j];
        brP[j] = s;
    }
}

// ---- t[n,h,c] = sum_d q[n,h*16+d]*WrP[c,h*16+d]; qb[n,h] = q[n,h]·brP[h] ----
__global__ __launch_bounds__(256) void compute_t_kernel(
    const float* __restrict__ q, const float* __restrict__ WrP,
    const float* __restrict__ brP,
    float* __restrict__ t, float* __restrict__ qb, int N) {
    __shared__ float swp[N_CENT * D_MODEL];
    __shared__ float sbp[D_MODEL];
    for (int i = threadIdx.x; i < N_CENT * D_MODEL; i += 256) swp[i] = WrP[i];
    for (int i = threadIdx.x; i < D_MODEL; i += 256) sbp[i] = brP[i];
    __syncthreads();

    int gid = blockIdx.x * 256 + threadIdx.x;           // (n, h)
    if (gid >= N * N_HEADS) return;
    int n = gid >> 2, h = gid & 3;
    float qh[HEAD_DIM];
    const float* qp = q + (size_t)n * 64 + h * 16;
#pragma unroll
    for (int i = 0; i < 16; ++i) qh[i] = qp[i];
    float qbv = 0.f;
#pragma unroll
    for (int i = 0; i < 16; ++i) qbv += qh[i] * sbp[h * 16 + i];
    qb[gid] = qbv;
    float* tp = t + (size_t)n * 128 + h * 32;
    for (int c = 0; c < N_CENT; ++c) {
        float s = 0.f;
#pragma unroll
        for (int i = 0; i < 16; ++i) s += qh[i] * swp[c * 64 + h * 16 + i];
        tp[c] = s;
    }
}

// ---- per-edge logits + leaky relu + atomic segment max ----
__global__ __launch_bounds__(256) void edge_logits_kernel(
    const int* __restrict__ src, const int* __restrict__ dst,
    const float* __restrict__ angle,
    const float* __restrict__ q, const float* __restrict__ k,
    const float* __restrict__ t, const float* __restrict__ qb,
    float* __restrict__ logit, unsigned* __restrict__ mkey, int E) {
    int e = blockIdx.x * 256 + threadIdx.x;
    if (e >= E) return;
    // software prefetch for a block-ahead edge's gather rows (global_prefetch_b8)
    int ep = e + 2048;
    if (ep < E) {
        __builtin_prefetch(k + (size_t)src[ep] * 64, 0, 0);
        __builtin_prefetch(q + (size_t)dst[ep] * 64, 0, 0);
        __builtin_prefetch(t + (size_t)dst[ep] * 128, 0, 0);
    }
    const int s = src[e], d = dst[e];
    const float ang = angle[e];
    float rbf[N_CENT];
#pragma unroll
    for (int c = 0; c < N_CENT; ++c) {
        float dx = ang - 0.1f * (float)c;
        rbf[c] = __expf(-RBF_G * dx * dx);
    }
    const float4* qp = (const float4*)(q + (size_t)d * 64);
    const float4* kp = (const float4*)(k + (size_t)s * 64);
    const float4* tp = (const float4*)(t + (size_t)d * 128);
#pragma unroll
    for (int h = 0; h < N_HEADS; ++h) {
        float acc = qb[(size_t)d * 4 + h];
#pragma unroll
        for (int i = 0; i < 4; ++i) {
            float4 qv = qp[h * 4 + i], kv = kp[h * 4 + i];
            acc += qv.x * kv.x + qv.y * kv.y + qv.z * kv.z + qv.w * kv.w;
        }
#pragma unroll
        for (int c4 = 0; c4 < 8; ++c4) {
            float4 tv = tp[h * 8 + c4];
            acc += rbf[c4 * 4 + 0] * tv.x + rbf[c4 * 4 + 1] * tv.y +
                   rbf[c4 * 4 + 2] * tv.z + rbf[c4 * 4 + 3] * tv.w;
        }
        float l = acc * 0.25f;                          // 1/sqrt(16)
        l = (l > 0.f) ? l : LEAKY_F * l;
        logit[(size_t)e * 4 + h] = l;
        atomicMax(&mkey[(size_t)d * 4 + h], f2key(l));
    }
}

// ---- p = exp(l - m[dst]); atomic segment sum (in-place over logit buffer) ----
__global__ __launch_bounds__(256) void edge_exp_kernel(
    const int* __restrict__ dst, float* __restrict__ att,
    const unsigned* __restrict__ mkey, float* __restrict__ z, int E) {
    int e = blockIdx.x * 256 + threadIdx.x;
    if (e >= E) return;
    const int d = dst[e];
#pragma unroll
    for (int h = 0; h < N_HEADS; ++h) {
        float m = key2f(mkey[(size_t)d * 4 + h]);
        float p = __expf(att[(size_t)e * 4 + h] - m);
        att[(size_t)e * 4 + h] = p;
        atomicAdd(&z[(size_t)d * 4 + h], p);
    }
}

__global__ void zinv_kernel(const float* __restrict__ z, float* __restrict__ zinv, int NH) {
    int i = blockIdx.x * 256 + threadIdx.x;
    if (i < NH) zinv[i] = 1.0f / (z[i] + 1e-9f);
}

// ---- one propagation hop: agg[dst] += att*zinv[dst] * feat[src]; 4 thr/edge ----
__global__ __launch_bounds__(256) void hop_kernel(
    const int* __restrict__ src, const int* __restrict__ dst,
    const float* __restrict__ att, const float* __restrict__ zinv,
    const float* __restrict__ feat, float* __restrict__ agg, int E) {
    int gid = blockIdx.x * 256 + threadIdx.x;
    if (gid >= E * N_HEADS) return;
    int e = gid >> 2, h = gid & 3;
    // prefetch the src row of an edge several waves ahead
    int ep = e + 512;
    if (ep < E) __builtin_prefetch(feat + (size_t)src[ep] * 64 + h * 16, 0, 0);
    const int s = src[e], d = dst[e];
    const float a = att[(size_t)e * 4 + h] * zinv[(size_t)d * 4 + h];
    const float4* fp = (const float4*)(feat + (size_t)s * 64 + h * 16);
    float* ap = agg + (size_t)d * 64 + h * 16;
#pragma unroll
    for (int i = 0; i < 4; ++i) {
        float4 f = fp[i];
        atomicAdd(ap + i * 4 + 0, a * f.x);
        atomicAdd(ap + i * 4 + 1, a * f.y);
        atomicAdd(ap + i * 4 + 2, a * f.z);
        atomicAdd(ap + i * 4 + 3, a * f.w);
    }
}

// ---- feat = ALPHA*v + (1-ALPHA)*agg, vectorized ----
__global__ void combine_kernel(const float* __restrict__ v,
                               const float* __restrict__ agg,
                               float* __restrict__ feat, int nel4) {
    int i = blockIdx.x * 256 + threadIdx.x;
    if (i >= nel4) return;
    float4 vv = ((const float4*)v)[i];
    float4 av = ((const float4*)agg)[i];
    float4 r;
    r.x = ALPHA_F * vv.x + (1.f - ALPHA_F) * av.x;
    r.y = ALPHA_F * vv.y + (1.f - ALPHA_F) * av.y;
    r.z = ALPHA_F * vv.z + (1.f - ALPHA_F) * av.z;
    r.w = ALPHA_F * vv.w + (1.f - ALPHA_F) * av.w;
    ((float4*)feat)[i] = r;
}

// =========================== host orchestration =============================
extern "C" void kernel_launch(void* const* d_in, const int* in_sizes, int n_in,
                              void* d_out, int out_size, void* d_ws, size_t ws_size,
                              hipStream_t stream) {
    const int*   src    = (const int*)d_in[0];
    const int*   dst    = (const int*)d_in[1];
    const float* angle  = (const float*)d_in[2];
    const float* h0     = (const float*)d_in[3];
    const float* rbf_W  = (const float*)d_in[4];
    const float* rbf_b  = (const float*)d_in[5];
    const float* Wq     = (const float*)d_in[6];
    const float* Wk     = (const float*)d_in[7];
    const float* Wv     = (const float*)d_in[8];
    const float* Wr     = (const float*)d_in[9];
    const float* Wo     = (const float*)d_in[10];

    const int E = in_sizes[0];
    const int N = in_sizes[3] / D_MODEL;

    // ----- scratch carve-out -----
    char*  ws  = (char*)d_ws;
    size_t off = 0;
    auto carve = [&](size_t bytes) -> void* {
        void* p = ws + off;
        off += (bytes + 255) & ~(size_t)255;
        return p;
    };
    float*    q    = (float*)carve((size_t)N * 64 * 4);
    float*    kbuf = (float*)carve((size_t)N * 64 * 4);
    float*    v    = (float*)carve((size_t)N * 64 * 4);
    float*    t    = (float*)carve((size_t)N * 128 * 4);
    float*    qb   = (float*)carve((size_t)N * 4 * 4);
    unsigned* mkey = (unsigned*)carve((size_t)N * 4 * 4);
    float*    z    = (float*)carve((size_t)N * 4 * 4);
    float*    zinv = (float*)carve((size_t)N * 4 * 4);
    float*    att  = (float*)carve((size_t)E * 4 * 4);
    float*    agg  = (float*)carve((size_t)N * 64 * 4);
    float*    feat = (float*)carve((size_t)N * 64 * 4);
    float*    htmp = (float*)carve((size_t)N * 64 * 4);
    float*    WrP  = (float*)carve((size_t)N_CENT * D_MODEL * 4);
    float*    brP  = (float*)carve((size_t)D_MODEL * 4);

    const int ntiles      = N / 16;                 // N = 100000 -> 6250 tiles
    const int gemm_blocks = (ntiles + 7) / 8;
    const int eblocks     = (E + 255) / 256;
    const int ehblocks    = (E * N_HEADS + 255) / 256;
    const int nhblocks    = (N * N_HEADS + 255) / 256;
    const int c4blocks    = (N * 16 + 255) / 256;   // N*64/4 float4s

    const float* hcur = h0;
    for (int l = 0; l < 2; ++l) {
        const size_t woff = (size_t)l * D_MODEL * D_MODEL;
        hipMemsetAsync(mkey, 0, (size_t)N * 4 * 4, stream);
        hipMemsetAsync(z,    0, (size_t)N * 4 * 4, stream);

        wrp_kernel<<<1, 256, 0, stream>>>(rbf_W, rbf_b, Wr + woff, WrP, brP);
        gemm3_kernel<<<gemm_blocks, 256, 0, stream>>>(hcur, ntiles,
            Wq + woff, Wk + woff, Wv + woff, q, kbuf, v);
        compute_t_kernel<<<nhblocks, 256, 0, stream>>>(q, WrP, brP, t, qb, N);
        edge_logits_kernel<<<eblocks, 256, 0, stream>>>(src, dst, angle,
            q, kbuf, t, qb, att, mkey, E);
        edge_exp_kernel<<<eblocks, 256, 0, stream>>>(dst, att, mkey, z, E);
        zinv_kernel<<<nhblocks, 256, 0, stream>>>(z, zinv, N * N_HEADS);

        // hop 1 (feat_in = v)
        hipMemsetAsync(agg, 0, (size_t)N * 64 * 4, stream);
        hop_kernel<<<ehblocks, 256, 0, stream>>>(src, dst, att, zinv, v, agg, E);
        combine_kernel<<<c4blocks, 256, 0, stream>>>(v, agg, feat, N * 16);
        // hop 2
        hipMemsetAsync(agg, 0, (size_t)N * 64 * 4, stream);
        hop_kernel<<<ehblocks, 256, 0, stream>>>(src, dst, att, zinv, feat, agg, E);
        combine_kernel<<<c4blocks, 256, 0, stream>>>(v, agg, feat, N * 16);

        float* outp = (l == 1) ? (float*)d_out : htmp;
        gemm1_kernel<<<gemm_blocks, 256, 0, stream>>>(feat, ntiles, Wo + woff, outp);
        hcur = htmp;
    }
}
